// SuperGlue_72069551227147
// MI455X (gfx1250) — matrix operand order repaired and checked by reading, not compiled, not run on hardware
//
#include <hip/hip_runtime.h>
#include <hip/hip_bf16.h>
#include <math.h>

// ---------------------------------------------------------------------------
// SuperGlue inference for gfx1250 (MI455X).  All GEMMs run through
// v_wmma_f32_16x16x32_f16 (f16 A/B, f32 accumulate).  Compute-bound workload
// (~175 GFLOP of GEMM on <50MB of L2-resident data).  GEMM: 128x128 block
// tile (8 wave32 waves x 8 WMMA tiles), compile-time trans variants,
// addrspace(1) b128 staging loads, LDS ping-pong (1 barrier/K-step),
// up-front fragment loads, global prefetch of the next K tile.
// ---------------------------------------------------------------------------

typedef _Float16 half4 __attribute__((ext_vector_type(4)));
typedef _Float16 half8 __attribute__((ext_vector_type(8)));
typedef _Float16 v16h  __attribute__((ext_vector_type(16)));
typedef float    v8f   __attribute__((ext_vector_type(8)));
typedef float    f32x4 __attribute__((ext_vector_type(4)));

__device__ inline v16h make_v16h(half8 lo, half8 hi) {
  return __builtin_shufflevector(lo, hi, 0,1,2,3,4,5,6,7,8,9,10,11,12,13,14,15);
}

// force GLOBAL (addrspace 1) b128 load instead of FLAT
__device__ inline f32x4 gload4(const float* p) {
  return *(const __attribute__((address_space(1))) f32x4*)p;
}

// ---------------------------------------------------------------------------
// Generic WMMA GEMM:  C[M,N] = alpha * op(A)[M,K] * op(B)[K,N]  (+bias,BN,ReLU)
//   TA: element(m,k) = A[k*lda + m]   (else A[m*lda + k])
//   TB: element(k,n) = B[n*ldb + k]   (else B[k*ldb + n])
// Block = 256 threads = 8 waves in a 4x2 grid; each wave owns an
// (TM/4)x(TN/2) sub-tile = MT x NT accumulators of 16x16.
// Requires M%TM==0, N%TN==0, K%32==0 (true for every call here).
// ---------------------------------------------------------------------------
template <int TM, int TN, int TA, int TB>
__global__ __launch_bounds__(256) void gemm_f16_wmma(
    const float* __restrict__ A, long lda, long aBatch,
    const float* __restrict__ B, long ldb, long bBatch,
    float* __restrict__ C, long ldc, long cBatch,
    int K, float alpha,
    const float* __restrict__ bias,
    const float* __restrict__ bng, const float* __restrict__ bnb,
    const float* __restrict__ bnm, const float* __restrict__ bnv,
    int relu)
{
  constexpr int MT  = TM / 64;            // per-wave row tiles (1 or 2)
  constexpr int NT  = TN / 32;            // per-wave col tiles (2 or 4)
  constexpr int LDT = 40;                 // LDS stride (halves), pad vs banks
  constexpr int APASS = (TM * 32) / 1024; // float4 staging passes, A
  constexpr int BPASS = (TN * 32) / 1024;
  constexpr int L2TM = (TM == 128) ? 7 : 6;
  constexpr int L2TN = (TN == 128) ? 7 : 6;

  __shared__ __align__(16) _Float16 As[2][TM * LDT];
  __shared__ __align__(16) _Float16 Bs[2][TN * LDT];

  const int t  = threadIdx.x;
  const int bm = blockIdx.y * TM;
  const int bn = blockIdx.x * TN;
  const int bz = blockIdx.z;
  const float* Ap = A + (long)bz * aBatch;
  const float* Bp = B + (long)bz * bBatch;
  float*       Cp = C + (long)bz * cBatch;

  const int lane = t & 31, wid = t >> 5;
  const int wr = wid & 3;          // wave row 0..3
  const int wc = wid >> 2;         // wave col 0..1
  const int l15 = lane & 15, lhi = lane >> 4;

  // ---- per-thread staging descriptors: one base + constant per-pass steps --
  const float* aCur; int aOff0; long aPassStep; int aLdsStep;
  if constexpr (!TA) {
    int i = t << 2; int m = i >> 5, k = i & 31;        // 4 consecutive k
    aCur = Ap + (long)(bm + m) * lda + k;
    aOff0 = m * LDT + k;
    aPassStep = 32 * lda;  aLdsStep = 32 * LDT;        // m += 32 per pass
  } else {
    int i = t << 2; int m = i & (TM - 1), k = i >> L2TM; // 4 consecutive m
    aCur = Ap + (long)k * lda + (bm + m);
    aOff0 = m * LDT + k;
    aPassStep = (1024 >> L2TM) * lda; aLdsStep = (1024 >> L2TM); // k advance
  }
  const long aKStep = TA ? 32 * lda : 32;

  const float* bCur; int bOff0; long bPassStep; int bLdsStep;
  if constexpr (!TB) {
    int i = t << 2; int n = i & (TN - 1), k = i >> L2TN; // 4 consecutive n
    bCur = Bp + (long)k * ldb + (bn + n);
    bOff0 = n * LDT + k;
    bPassStep = (1024 >> L2TN) * ldb; bLdsStep = (1024 >> L2TN);
  } else {
    int i = t << 2; int k = i & 31, n = i >> 5;          // 4 consecutive k
    bCur = Bp + (long)(bn + n) * ldb + k;
    bOff0 = n * LDT + k;
    bPassStep = 32 * ldb;  bLdsStep = 32 * LDT;          // n += 32 per pass
  }
  const long bKStep = TB ? 32 : 32 * ldb;

  v8f acc[MT][NT] = {};
  int pb = 0;

  for (int k0 = 0; k0 < K; k0 += 32) {
    // ---- batch all staging loads first (global_load_b128, many in flight) --
    f32x4 aReg[APASS];
    #pragma unroll
    for (int p = 0; p < APASS; ++p) aReg[p] = gload4(aCur + p * aPassStep);
    f32x4 bReg[BPASS];
    #pragma unroll
    for (int p = 0; p < BPASS; ++p) bReg[p] = gload4(bCur + p * bPassStep);
    aCur += aKStep; bCur += bKStep;
    if (k0 + 32 < K) {                 // prefetch next K tile
      __builtin_prefetch(aCur, 0, 1);
      __builtin_prefetch(bCur, 0, 1);
    }

    // ---- convert + store to LDS ----
    #pragma unroll
    for (int p = 0; p < APASS; ++p) {
      f32x4 f = aReg[p];
      if constexpr (!TA) {
        half4 h = { (_Float16)f[0], (_Float16)f[1], (_Float16)f[2], (_Float16)f[3] };
        *(half4*)&As[pb][aOff0 + p * aLdsStep] = h;
      } else {
        _Float16* d = &As[pb][aOff0 + p * aLdsStep];
        d[0 * LDT] = (_Float16)f[0]; d[1 * LDT] = (_Float16)f[1];
        d[2 * LDT] = (_Float16)f[2]; d[3 * LDT] = (_Float16)f[3];
      }
    }
    #pragma unroll
    for (int p = 0; p < BPASS; ++p) {
      f32x4 f = bReg[p];
      if constexpr (!TB) {
        _Float16* d = &Bs[pb][bOff0 + p * bLdsStep];
        d[0 * LDT] = (_Float16)f[0]; d[1 * LDT] = (_Float16)f[1];
        d[2 * LDT] = (_Float16)f[2]; d[3 * LDT] = (_Float16)f[3];
      } else {
        half4 h = { (_Float16)f[0], (_Float16)f[1], (_Float16)f[2], (_Float16)f[3] };
        *(half4*)&Bs[pb][bOff0 + p * bLdsStep] = h;
      }
    }
    __syncthreads();

    // ---- load all fragments up front, then 8 uninterrupted WMMAs ----
    // A frag (16x32): lane l<16 row M=l holds K 0-7,16-23; lanes 16-31 K 8-15,24-31
    v16h af[MT];
    #pragma unroll
    for (int mt = 0; mt < MT; ++mt) {
      int arow = wr * (MT * 16) + mt * 16 + l15;
      half8 alo = *(const half8*)&As[pb][arow * LDT + lhi * 8];
      half8 ahi = *(const half8*)&As[pb][arow * LDT + lhi * 8 + 16];
      af[mt] = make_v16h(alo, ahi);
    }
    v16h bf[NT];
    #pragma unroll
    for (int nt = 0; nt < NT; ++nt) {
      int bcol = wc * (NT * 16) + nt * 16 + l15;
      half8 blo = *(const half8*)&Bs[pb][bcol * LDT + lhi * 8];
      half8 bhi = *(const half8*)&Bs[pb][bcol * LDT + lhi * 8 + 16];
      bf[nt] = make_v16h(blo, bhi);
    }
    #pragma unroll
    for (int nt = 0; nt < NT; ++nt)
      #pragma unroll
      for (int mt = 0; mt < MT; ++mt)
        acc[mt][nt] = __builtin_amdgcn_wmma_f32_16x16x32_f16(
            false, af[mt], false, bf[nt], (short)0, acc[mt][nt], false, false);

    pb ^= 1;   // ping-pong: single barrier per K-step is sufficient
  }

  // ---- epilogue: C layout VGPR r -> M = r + (lane>=16 ? 8 : 0), N = lane&15
  #pragma unroll
  for (int mt = 0; mt < MT; ++mt) {
    #pragma unroll
    for (int r = 0; r < 8; ++r) {
      int mrow = bm + wr * (MT * 16) + mt * 16 + r + lhi * 8;
      float bval = bias ? bias[mrow] : 0.f;
      float s = 1.f, sh = 0.f;
      if (bng) { s = bng[mrow] * rsqrtf(bnv[mrow] + 1e-5f); sh = bnb[mrow] - bnm[mrow] * s; }
      #pragma unroll
      for (int nt = 0; nt < NT; ++nt) {
        float val = acc[mt][nt][r] * alpha + bval;
        if (bng)  val = val * s + sh;
        if (relu) val = fmaxf(val, 0.f);
        Cp[(long)mrow * ldc + (bn + wc * (NT * 16) + nt * 16 + l15)] = val;
      }
    }
  }
}

// ---------------------------------------------------------------------------
// Keypoint encoder MLP (3->32->64->128->256, BN+ReLU between) + desc residual.
// ---------------------------------------------------------------------------
__global__ __launch_bounds__(256) void kenc_kernel(
    const float* __restrict__ kpts, const float* __restrict__ kscores,
    const float* __restrict__ desc,
    const float* w0, const float* w1, const float* w2, const float* w3,
    const float* b0, const float* b1, const float* b2, const float* b3,
    const float* g0, const float* g1, const float* g2,
    const float* bb0, const float* bb1, const float* bb2,
    const float* mm0, const float* mm1, const float* mm2,
    const float* vv0, const float* vv1, const float* vv2,
    float* __restrict__ xout)
{
  const int n = blockIdx.x, b = blockIdx.y, t = threadIdx.x;
  __shared__ float bufA[256];
  __shared__ float bufB[256];
  if (t == 0) {
    bufA[0] = (kpts[((long)b * 1024 + n) * 2 + 0] - 320.f) / 448.f;  // (x-W/2)/(0.7*max)
    bufA[1] = (kpts[((long)b * 1024 + n) * 2 + 1] - 180.f) / 448.f;
    bufA[2] = kscores[(long)b * 1024 + n];
  }
  __syncthreads();
  if (t < 32) {
    float a = b0[t];
    for (int c = 0; c < 3; ++c) a += w0[t * 3 + c] * bufA[c];
    float s = g0[t] * rsqrtf(vv0[t] + 1e-5f);
    bufB[t] = fmaxf((a - mm0[t]) * s + bb0[t], 0.f);
  }
  __syncthreads();
  if (t < 64) {
    float a = b1[t];
    for (int c = 0; c < 32; ++c) a += w1[t * 32 + c] * bufB[c];
    float s = g1[t] * rsqrtf(vv1[t] + 1e-5f);
    bufA[t] = fmaxf((a - mm1[t]) * s + bb1[t], 0.f);
  }
  __syncthreads();
  if (t < 128) {
    float a = b2[t];
    for (int c = 0; c < 64; ++c) a += w2[t * 64 + c] * bufA[c];
    float s = g2[t] * rsqrtf(vv2[t] + 1e-5f);
    bufB[t] = fmaxf((a - mm2[t]) * s + bb2[t], 0.f);
  }
  __syncthreads();
  {
    float a = b3[t];
    for (int c = 0; c < 128; ++c) a += w3[t * 128 + c] * bufB[c];
    long o = (long)b * 262144 + (long)t * 1024 + n;
    xout[o] = desc[o] + a;
  }
}

// ---------------------------------------------------------------------------
// Row softmax (in place).
// ---------------------------------------------------------------------------
__global__ __launch_bounds__(256) void softmax_rows(float* __restrict__ X,
                                                    long rowStride, long batchStride,
                                                    int cols)
{
  const int row = blockIdx.x, b = blockIdx.y, t = threadIdx.x;
  float* p = X + (long)b * batchStride + (long)row * rowStride;
  __shared__ float red[256];
  float m = -3.4e38f;
  for (int j = t; j < cols; j += 256) m = fmaxf(m, p[j]);
  red[t] = m; __syncthreads();
  for (int s = 128; s > 0; s >>= 1) { if (t < s) red[t] = fmaxf(red[t], red[t + s]); __syncthreads(); }
  m = red[0]; __syncthreads();
  float sum = 0.f;
  for (int j = t; j < cols; j += 256) { float e = __expf(p[j] - m); p[j] = e; sum += e; }
  red[t] = sum; __syncthreads();
  for (int s = 128; s > 0; s >>= 1) { if (t < s) red[t] += red[t + s]; __syncthreads(); }
  float inv = 1.f / red[0];
  for (int j = t; j < cols; j += 256) p[j] *= inv;
}

// ---------------------------------------------------------------------------
// Sinkhorn updates (log domain).  Z0 is (B,1025,1025).
// ---------------------------------------------------------------------------
__global__ __launch_bounds__(256) void sinkhorn_u_kernel(const float* __restrict__ Z0,
                                                         const float* __restrict__ v,
                                                         float* __restrict__ u)
{
  const int i = blockIdx.x, b = blockIdx.y, t = threadIdx.x;
  const float* z  = Z0 + (long)b * 1050625 + (long)i * 1025;
  const float* vv = v + b * 1025;
  __shared__ float red[256];
  float m = -3.4e38f;
  for (int j = t; j < 1025; j += 256) m = fmaxf(m, z[j] + vv[j]);
  red[t] = m; __syncthreads();
  for (int s = 128; s > 0; s >>= 1) { if (t < s) red[t] = fmaxf(red[t], red[t + s]); __syncthreads(); }
  m = red[0]; __syncthreads();
  float sum = 0.f;
  for (int j = t; j < 1025; j += 256) sum += __expf(z[j] + vv[j] - m);
  red[t] = sum; __syncthreads();
  for (int s = 128; s > 0; s >>= 1) { if (t < s) red[t] += red[t + s]; __syncthreads(); }
  if (t == 0) {
    const float norm = -logf(2049.f);
    float lmu = (i < 1024) ? norm : (logf(1024.f) + norm);
    u[b * 1025 + i] = lmu - (logf(red[0]) + m);
  }
}

__global__ __launch_bounds__(256) void sinkhorn_v_kernel(const float* __restrict__ Z0,
                                                         const float* __restrict__ u,
                                                         float* __restrict__ v)
{
  const int j = blockIdx.x, b = blockIdx.y, t = threadIdx.x;
  const float* z  = Z0 + (long)b * 1050625 + j;
  const float* uu = u + b * 1025;
  __shared__ float red[256];
  float m = -3.4e38f;
  for (int i = t; i < 1025; i += 256) m = fmaxf(m, z[(long)i * 1025] + uu[i]);
  red[t] = m; __syncthreads();
  for (int s = 128; s > 0; s >>= 1) { if (t < s) red[t] = fmaxf(red[t], red[t + s]); __syncthreads(); }
  m = red[0]; __syncthreads();
  float sum = 0.f;
  for (int i = t; i < 1025; i += 256) sum += __expf(z[(long)i * 1025] + uu[i] - m);
  red[t] = sum; __syncthreads();
  for (int s = 128; s > 0; s >>= 1) { if (t < s) red[t] += red[t + s]; __syncthreads(); }
  if (t == 0) {
    const float norm = -logf(2049.f);
    float lnu = (j < 1024) ? norm : (logf(1024.f) + norm);
    v[b * 1025 + j] = lnu - (logf(red[0]) + m);
  }
}

// ---------------------------------------------------------------------------
// Row max + argmax over inner (1024x1024) of Z = Z0 + u + v - norm.
// ---------------------------------------------------------------------------
__global__ __launch_bounds__(256) void rowmax_kernel(const float* __restrict__ Z0,
                                                     const float* __restrict__ u,
                                                     const float* __restrict__ v,
                                                     float* __restrict__ max0v,
                                                     int* __restrict__ idx0)
{
  const int i = blockIdx.x, b = blockIdx.y, t = threadIdx.x;
  const float* z = Z0 + (long)b * 1050625 + (long)i * 1025;
  const float ui = u[b * 1025 + i];
  const float norm = -logf(2049.f);
  __shared__ float sv[256]; __shared__ int si[256];
  float best = -3.4e38f; int bj = 0;
  for (int j = t; j < 1024; j += 256) {
    float val = z[j] + ui + v[b * 1025 + j] - norm;
    if (val > best) { best = val; bj = j; }
  }
  sv[t] = best; si[t] = bj; __syncthreads();
  for (int s = 128; s > 0; s >>= 1) {
    if (t < s) {
      if (sv[t + s] > sv[t] || (sv[t + s] == sv[t] && si[t + s] < si[t])) {
        sv[t] = sv[t + s]; si[t] = si[t + s];
      }
    }
    __syncthreads();
  }
  if (t == 0) { max0v[(long)b * 1024 + i] = sv[0]; idx0[(long)b * 1024 + i] = si[0]; }
}

__global__ __launch_bounds__(256) void colargmax_kernel(const float* __restrict__ Z0,
                                                        const float* __restrict__ u,
                                                        int* __restrict__ idx1)
{
  const int j = blockIdx.x, b = blockIdx.y, t = threadIdx.x;
  const float* z = Z0 + (long)b * 1050625 + j;
  __shared__ float sv[256]; __shared__ int si[256];
  float best = -3.4e38f; int bi = 0;
  for (int i = t; i < 1024; i += 256) {
    float val = z[(long)i * 1025] + u[b * 1025 + i];  // +v[j]-norm constant per column
    if (val > best) { best = val; bi = i; }
  }
  sv[t] = best; si[t] = bi; __syncthreads();
  for (int s = 128; s > 0; s >>= 1) {
    if (t < s) {
      if (sv[t + s] > sv[t] || (sv[t + s] == sv[t] && si[t + s] < si[t])) {
        sv[t] = sv[t + s]; si[t] = si[t + s];
      }
    }
    __syncthreads();
  }
  if (t == 0) idx1[(long)b * 1024 + j] = si[0];
}

// ---------------------------------------------------------------------------
// Matching epilogues.
// ---------------------------------------------------------------------------
__global__ __launch_bounds__(256) void match0_kernel(const float* __restrict__ max0v,
                                                     const int* __restrict__ idx0,
                                                     const int* __restrict__ idx1,
                                                     float* __restrict__ out_i0,
                                                     float* __restrict__ out_msc0,
                                                     int* __restrict__ valid0)
{
  const int n = blockIdx.x * 256 + threadIdx.x, b = blockIdx.y;
  if (n >= 1024) return;
  long o = (long)b * 1024 + n;
  int j = idx0[o];
  bool mut = (idx1[(long)b * 1024 + j] == n);
  float msc = mut ? __expf(max0v[o]) : 0.f;
  bool val = mut && (msc > 0.f);
  out_i0[o]   = val ? (float)j : -1.f;
  out_msc0[o] = msc;
  valid0[o]   = val ? 1 : 0;
}

__global__ __launch_bounds__(256) void match1_kernel(const int* __restrict__ idx0,
                                                     const int* __restrict__ idx1,
                                                     const float* __restrict__ msc0,
                                                     const int* __restrict__ valid0,
                                                     float* __restrict__ out_i1,
                                                     float* __restrict__ out_msc1)
{
  const int j = blockIdx.x * 256 + threadIdx.x, b = blockIdx.y;
  if (j >= 1024) return;
  long o = (long)b * 1024 + j;
  int i = idx1[o];
  bool mut = (idx0[(long)b * 1024 + i] == j);
  float m1 = mut ? msc0[(long)b * 1024 + i] : 0.f;
  bool v1 = mut && (valid0[(long)b * 1024 + i] != 0);
  out_i1[o]   = v1 ? (float)i : -1.f;
  out_msc1[o] = m1;
}

// ---------------------------------------------------------------------------
// Small helpers.
// ---------------------------------------------------------------------------
__global__ __launch_bounds__(256) void concat_kernel(const float* __restrict__ x,
                                                     const float* __restrict__ msg,
                                                     float* __restrict__ cat)
{
  long i = (long)blockIdx.x * 256 + threadIdx.x;
  if (i >= 2l * 512 * 1024) return;
  long b = i >> 19;
  long r = i & 524287;
  long c = r >> 10, n = r & 1023;
  cat[i] = (c < 256) ? x[b * 262144 + c * 1024 + n]
                     : msg[b * 262144 + (c - 256) * 1024 + n];
}

__global__ __launch_bounds__(256) void add_kernel(float* __restrict__ x,
                                                  const float* __restrict__ d, long nel)
{
  long i = (long)blockIdx.x * 256 + threadIdx.x;
  if (i < nel) x[i] += d[i];
}

__global__ __launch_bounds__(256) void zero_kernel(float* __restrict__ p, long nel)
{
  long i = (long)blockIdx.x * 256 + threadIdx.x;
  if (i < nel) p[i] = 0.f;
}

__global__ __launch_bounds__(256) void fill_bins_kernel(float* __restrict__ Z0,
                                                        const float* __restrict__ bin)
{
  const int t = blockIdx.x * 256 + threadIdx.x, b = blockIdx.y;
  const float a = *bin;
  if (t < 1025)      Z0[(long)b * 1050625 + (long)t * 1025 + 1024] = a;       // last col + corner
  else if (t < 2049) Z0[(long)b * 1050625 + 1024l * 1025 + (t - 1025)] = a;   // last row
}

// ---------------------------------------------------------------------------
// Host orchestration.
// ---------------------------------------------------------------------------
extern "C" void kernel_launch(void* const* d_in, const int* in_sizes, int n_in,
                              void* d_out, int out_size, void* d_ws, size_t ws_size,
                              hipStream_t stream)
{
  (void)in_sizes; (void)n_in; (void)out_size; (void)ws_size;

  // Input order: top-level dict order; nested dicts flattened in sorted-key
  // order (jax pytree), lists element-by-element.
  const float* kpts0   = (const float*)d_in[0];
  const float* scores0 = (const float*)d_in[1];
  const float* desc0   = (const float*)d_in[2];
  const float* kpts1   = (const float*)d_in[3];
  const float* scores1 = (const float*)d_in[4];
  const float* desc1   = (const float*)d_in[5];
  // kenc_params: b[0..3], bn_b[0..2], bn_g[0..2], bn_m[0..2], bn_v[0..2], w[0..3]
  const float* kb[4]  = {(const float*)d_in[6],  (const float*)d_in[7],
                         (const float*)d_in[8],  (const float*)d_in[9]};
  const float* kbb[3] = {(const float*)d_in[10], (const float*)d_in[11], (const float*)d_in[12]};
  const float* kbg[3] = {(const float*)d_in[13], (const float*)d_in[14], (const float*)d_in[15]};
  const float* kbm[3] = {(const float*)d_in[16], (const float*)d_in[17], (const float*)d_in[18]};
  const float* kbv[3] = {(const float*)d_in[19], (const float*)d_in[20], (const float*)d_in[21]};
  const float* kw[4]  = {(const float*)d_in[22], (const float*)d_in[23],
                         (const float*)d_in[24], (const float*)d_in[25]};
  // gnn_params sorted keys: bn_b, bn_g, bn_m, bn_v, merge_b, merge_w,
  //                         mlp_b1, mlp_b2, mlp_w1, mlp_w2, proj_b, proj_w
  const float* g_bnb    = (const float*)d_in[26];
  const float* g_bng    = (const float*)d_in[27];
  const float* g_bnm    = (const float*)d_in[28];
  const float* g_bnv    = (const float*)d_in[29];
  const float* g_mergeb = (const float*)d_in[30];
  const float* g_mergew = (const float*)d_in[31];
  const float* g_mlpb1  = (const float*)d_in[32];
  const float* g_mlpb2  = (const float*)d_in[33];
  const float* g_mlpw1  = (const float*)d_in[34];
  const float* g_mlpw2  = (const float*)d_in[35];
  const float* g_projb  = (const float*)d_in[36];
  const float* g_projw  = (const float*)d_in[37];
  const float* final_w  = (const float*)d_in[38];
  const float* final_b  = (const float*)d_in[39];
  const float* bin_sc   = (const float*)d_in[40];

  float* out = (float*)d_out;
  float* out_i0   = out + 0;
  float* out_i1   = out + 2048;
  float* out_msc0 = out + 4096;
  float* out_msc1 = out + 6144;

  // workspace carve-up (floats)
  size_t off = 0;
  float* base = (float*)d_ws;
  auto alloc = [&](size_t nf) { float* p = base + off; off += nf; return p; };
  const long DN = 262144;               // 256*1024 per batch
  float* x0     = alloc(2 * DN);
  float* x1     = alloc(2 * DN);
  float* qb     = alloc(2 * DN);
  float* kbuf   = alloc(2 * DN);
  float* vbuf   = alloc(2 * DN);
  float* attn   = alloc(2 * DN);
  float* msg    = alloc(2 * DN);
  float* cat    = alloc(2 * 524288);
  float* hbuf   = alloc(2 * 524288);
  float* delta0 = alloc(2 * DN);
  float* delta1 = alloc(2 * DN);
  float* S      = alloc(2l * 1024 * 1024);
  float* Z0     = alloc(2l * 1025 * 1025);
  float* u      = alloc(2 * 1025);
  float* v      = alloc(2 * 1025);   // contiguous with u
  float* max0v  = alloc(2048);
  int*   idx0   = (int*)alloc(2048);
  int*   idx1   = (int*)alloc(2048);
  int*   valid0 = (int*)alloc(2048);

  // dispatch over (tile size, transA, transB); only 4 combos are used
  auto gemm = [&](const float* A, long lda, int tA, long aB,
                  const float* B, long ldb, int tB, long bB,
                  float* C, long ldc, long cB,
                  int M, int N, int K, float alpha,
                  const float* bias,
                  const float* bng, const float* bnb,
                  const float* bnm, const float* bnv, int relu) {
    if ((M & 127) == 0 && (N & 127) == 0) {
      dim3 grid(N / 128, M / 128, 2);
      if (tA)
        gemm_f16_wmma<128, 128, 1, 0><<<grid, 256, 0, stream>>>(
            A, lda, aB, B, ldb, bB, C, ldc, cB, K, alpha, bias, bng, bnb, bnm, bnv, relu);
      else
        gemm_f16_wmma<128, 128, 0, 0><<<grid, 256, 0, stream>>>(
            A, lda, aB, B, ldb, bB, C, ldc, cB, K, alpha, bias, bng, bnb, bnm, bnv, relu);
    } else {
      dim3 grid(N / 64, M / 64, 2);
      if (tB)
        gemm_f16_wmma<64, 64, 0, 1><<<grid, 256, 0, stream>>>(
            A, lda, aB, B, ldb, bB, C, ldc, cB, K, alpha, bias, bng, bnb, bnm, bnv, relu);
      else
        gemm_f16_wmma<64, 64, 0, 0><<<grid, 256, 0, stream>>>(
            A, lda, aB, B, ldb, bB, C, ldc, cB, K, alpha, bias, bng, bnb, bnm, bnv, relu);
    }
  };
  const float* NUL = nullptr;

  // ---- keypoint encoder + residual -> x0, x1 ----
  kenc_kernel<<<dim3(1024, 2), 256, 0, stream>>>(
      kpts0, scores0, desc0, kw[0], kw[1], kw[2], kw[3], kb[0], kb[1], kb[2], kb[3],
      kbg[0], kbg[1], kbg[2], kbb[0], kbb[1], kbb[2], kbm[0], kbm[1], kbm[2],
      kbv[0], kbv[1], kbv[2], x0);
  kenc_kernel<<<dim3(1024, 2), 256, 0, stream>>>(
      kpts1, scores1, desc1, kw[0], kw[1], kw[2], kw[3], kb[0], kb[1], kb[2], kb[3],
      kbg[0], kbg[1], kbg[2], kbb[0], kbb[1], kbb[2], kbm[0], kbm[1], kbm[2],
      kbv[0], kbv[1], kbv[2], x1);

  // ---- 18 GNN layers (self, cross alternating) ----
  for (int i = 0; i < 18; ++i) {
    const bool cross = (i & 1);
    const float* pw = g_projw + (long)i * 3 * 65536;
    const float* pb = g_projb + (long)i * 3 * 256;
    for (int img = 0; img < 2; ++img) {
      float* x    = img ? x1 : x0;
      float* src  = cross ? (img ? x0 : x1) : x;
      float* dlt  = img ? delta1 : delta0;
      // q,k,v projections (f16 WMMA)
      gemm(pw,          256, 0, 0, x,   1024, 0, DN, qb,   1024, DN, 256, 1024, 256, 1.f, pb,       NUL,NUL,NUL,NUL, 0);
      gemm(pw + 65536,  256, 0, 0, src, 1024, 0, DN, kbuf, 1024, DN, 256, 1024, 256, 1.f, pb + 256, NUL,NUL,NUL,NUL, 0);
      gemm(pw + 131072, 256, 0, 0, src, 1024, 0, DN, vbuf, 1024, DN, 256, 1024, 256, 1.f, pb + 512, NUL,NUL,NUL,NUL, 0);
      // attention per head: channel c = 4*dd + h
      for (int h = 0; h < 4; ++h) {
        // S[n,m] = (1/8) * sum_dd q[4dd+h][n] * k[4dd+h][m]   (transA gather)
        gemm(qb + h * 1024, 4096, 1, DN, kbuf + h * 1024, 4096, 0, DN,
             S, 1024, 1048576, 1024, 1024, 64, 0.125f, NUL, NUL,NUL,NUL,NUL, 0);
        softmax_rows<<<dim3(1024, 2), 256, 0, stream>>>(S, 1024, 1048576, 1024);
        // O[dd,n] = sum_m V_h[dd,m] * P[n,m]   (transB)  -> 64x64 tile variant
        gemm(vbuf + h * 1024, 4096, 0, DN, S, 1024, 1, 1048576,
             attn + h * 1024, 4096, DN, 64, 1024, 1024, 1.f, NUL, NUL,NUL,NUL,NUL, 0);
      }
      // merge
      gemm(g_mergew + (long)i * 65536, 256, 0, 0, attn, 1024, 0, DN,
           msg, 1024, DN, 256, 1024, 256, 1.f, g_mergeb + (long)i * 256, NUL,NUL,NUL,NUL, 0);
      // cat = [x ; msg]
      concat_kernel<<<dim3(4096), 256, 0, stream>>>(x, msg, cat);
      // mlp1 (+BN+ReLU fused)
      gemm(g_mlpw1 + (long)i * 262144, 512, 0, 0, cat, 1024, 0, 524288,
           hbuf, 1024, 524288, 512, 1024, 512, 1.f, g_mlpb1 + (long)i * 512,
           g_bng + (long)i * 512, g_bnb + (long)i * 512,
           g_bnm + (long)i * 512, g_bnv + (long)i * 512, 1);
      // mlp2 -> delta
      gemm(g_mlpw2 + (long)i * 131072, 512, 0, 0, hbuf, 1024, 0, 524288,
           dlt, 1024, DN, 256, 1024, 512, 1.f, g_mlpb2 + (long)i * 256, NUL,NUL,NUL,NUL, 0);
    }
    add_kernel<<<dim3(2048), 256, 0, stream>>>(x0, delta0, 2 * DN);
    add_kernel<<<dim3(2048), 256, 0, stream>>>(x1, delta1, 2 * DN);
  }

  // ---- final projections + match scores ----
  gemm(final_w, 256, 0, 0, x0, 1024, 0, DN, qb,   1024, DN, 256, 1024, 256, 1.f, final_b, NUL,NUL,NUL,NUL, 0);
  gemm(final_w, 256, 0, 0, x1, 1024, 0, DN, kbuf, 1024, DN, 256, 1024, 256, 1.f, final_b, NUL,NUL,NUL,NUL, 0);
  // scores[n,m] = (1/16) sum_d m0[d,n] m1[d,m]  -> write into Z0 inner (ldc=1025)
  gemm(qb, 1024, 1, DN, kbuf, 1024, 0, DN, Z0, 1025, 1050625,
       1024, 1024, 256, 1.f / 16.f, NUL, NUL,NUL,NUL,NUL, 0);
  fill_bins_kernel<<<dim3(9, 2), 256, 0, stream>>>(Z0, bin_sc);

  // ---- Sinkhorn (20 iterations, log domain) ----
  zero_kernel<<<dim3(17), 256, 0, stream>>>(u, 4100);  // u and v contiguous
  for (int it = 0; it < 20; ++it) {
    sinkhorn_u_kernel<<<dim3(1025, 2), 256, 0, stream>>>(Z0, v, u);
    sinkhorn_v_kernel<<<dim3(1025, 2), 256, 0, stream>>>(Z0, u, v);
  }

  // ---- argmax + mutual matching ----
  rowmax_kernel<<<dim3(1024, 2), 256, 0, stream>>>(Z0, u, v, max0v, idx0);
  colargmax_kernel<<<dim3(1024, 2), 256, 0, stream>>>(Z0, u, idx1);
  match0_kernel<<<dim3(4, 2), 256, 0, stream>>>(max0v, idx0, idx1, out_i0, out_msc0, valid0);
  match1_kernel<<<dim3(4, 2), 256, 0, stream>>>(idx0, idx1, out_msc0, valid0, out_i1, out_msc1);
}
